// MHA_18313740550637
// MI455X (gfx1250) — matrix operand order repaired and checked by reading, not compiled, not run on hardware
//
#include <hip/hip_runtime.h>
#include <hip/hip_bf16.h>

typedef __attribute__((ext_vector_type(16))) _Float16 v16h;
typedef __attribute__((ext_vector_type(8)))  _Float16 v8h;
typedef __attribute__((ext_vector_type(8)))  float    v8f;

typedef unsigned int       u32;
typedef unsigned long long u64;
typedef u32 tdm_u32x4 __attribute__((ext_vector_type(4)));
typedef int tdm_i32x8 __attribute__((ext_vector_type(8)));
typedef int tdm_i32x4 __attribute__((ext_vector_type(4)));

#define HIDDEN 1024
#define SEQ    2048
#define BATCH  2
#define HEADS  16
#define HDIM   64
#define MTOT   (BATCH * SEQ)   // 4096

static __device__ __forceinline__ v8f wmma_f16(v16h a, v16h b, v8f c) {
  return __builtin_amdgcn_wmma_f32_16x16x32_f16(false, a, false, b, (short)0, c,
                                                false, false);
}

// ---------------- Tensor Data Mover (D# per CDNA5 ISA) -------
// Loop-invariant group-1 descriptor (dims / strides / data_size=2B).
static __device__ __forceinline__ tdm_i32x8 tdm_desc_g1(
    u32 tile_d0, u32 tile_d1, u32 tensor_d0, u32 tensor_d1, u64 stride_d0) {
  u64 w0 = (1ull << 16)                                 // data_size = 1 -> 2B
         | ((u64)(tensor_d0 & 0xFFFFu) << 48);          // tensor_dim0[15:0]
  u64 w1 = ((u64)tensor_d0 >> 16)                       // tensor_dim0[31:16]
         | ((u64)tensor_d1 << 16)                       // tensor_dim1 @ bits 111:80
         | ((u64)(tile_d0 & 0xFFFFu) << 48);            // tile_dim0   @ bits 127:112
  u64 w2 = (u64)(tile_d1 & 0xFFFFu)                     // tile_dim1   @ bits 143:128
         | ((stride_d0 & 0xFFFFFFFFull) << 32);         // dim0_stride[31:0] @ 191:160
  u64 w3 = (stride_d0 >> 32) & 0xFFFFull;               // dim0_stride[47:32]
  tdm_i32x8 g1;
  g1[0] = (int)w0; g1[1] = (int)(w0 >> 32);
  g1[2] = (int)w1; g1[3] = (int)(w1 >> 32);
  g1[4] = (int)w2; g1[5] = (int)(w2 >> 32);
  g1[6] = (int)w3; g1[7] = (int)(w3 >> 32);
  return g1;
}

// Per-issue group 0 (LDS byte offset + global tile address) + issue.
static __device__ __forceinline__ void tdm_issue(u32 lds_off, const void* gaddr,
                                                 tdm_i32x8 g1) {
  u64 ga = (u64)(uintptr_t)gaddr;
  tdm_u32x4 g0;
  g0[0] = 1u;                                           // count=1 (user mode)
  g0[1] = lds_off;                                      // lds_addr (bytes)
  g0[2] = (u32)ga;                                      // global_addr[31:0]
  g0[3] = (u32)((ga >> 32) & 0x1FFFFFFu) | (2u << 30);  // addr[56:32] | type=2
  tdm_i32x4 z4 = {0, 0, 0, 0};
  tdm_i32x8 z8 = {0, 0, 0, 0, 0, 0, 0, 0};
  __builtin_amdgcn_tensor_load_to_lds(g0, g1, z4, z4, z8, 0);  // 6-arg form
}

// ---------------- WMMA fragment loaders (global or LDS pointers) --------
// A-fragment 16x32 f16 from row-major [rows, ld] at (row0, k0).
// Halves written through v8h views so loads land directly in the 8-VGPR tuple.
static __device__ __forceinline__ v16h load_a(const _Float16* __restrict__ base,
                                              int row0, int ld, int k0, int lane) {
  int r = lane & 15, hi = (lane >> 4) & 1;
  const _Float16* p = base + (size_t)(row0 + r) * ld + k0 + hi * 8;
  v16h out;
  v8h* oh = (v8h*)&out;
  oh[0] = *(const v8h*)(p);         // K = k0 + hi*8 .. +7
  oh[1] = *(const v8h*)(p + 16);    // K = k0 + 16 + hi*8 .. +7
  return out;
}

// B-fragment 32x16: column n = row (col0+n) of row-major [N, ld] (K contiguous).
static __device__ __forceinline__ v16h load_b(const _Float16* __restrict__ base,
                                              int col0, int ld, int k0, int lane) {
  int c = lane & 15, hi = (lane >> 4) & 1;
  const _Float16* p = base + (size_t)(col0 + c) * ld + k0 + hi * 8;
  v16h out;
  v8h* oh = (v8h*)&out;
  oh[0] = *(const v8h*)(p);
  oh[1] = *(const v8h*)(p + 16);
  return out;
}

__global__ void __launch_bounds__(256)
cvt_f32_f16(const float* __restrict__ in, _Float16* __restrict__ out, int n) {
  int i = blockIdx.x * blockDim.x + threadIdx.x;
  if (i < n) out[i] = (_Float16)in[i];
}

// ---------------- Tiled GEMM: Y = X[4096,1024] * W[1024,1024]^T + bias ---
// Block: 128x128 tile, 8 waves (4x2 of 32x64), TDM double-buffered K=32 steps.
// mode 0: fp32 [M,N]   mode 1: f16 [B,H,S,Dh]   mode 3: f16 [B,H,Dh,S]
#define BM 128
#define BN 128
#define BK 32
#define TILE_ELTS (BM * BK)   // 4096 f16 = 8 KB

__global__ void __launch_bounds__(256)
gemm_f16(const _Float16* __restrict__ X, const _Float16* __restrict__ W,
         const float* __restrict__ bias, void* __restrict__ Y, int mode) {
  // single shared block => starts at LDS offset 0
  // layout: A buf0 | A buf1 | B buf0 | B buf1 (8 KB each)
  __shared__ __align__(16) _Float16 smem[4 * TILE_ELTS];  // 32 KB

  int lane = threadIdx.x & 31;
  int wave = threadIdx.x >> 5;
  int mw = wave & 3;                 // 4 row-slots of 32
  int nw = wave >> 2;                // 2 col-slots of 64
  int bm = (blockIdx.x & 31) * BM;   // 32 M-tiles
  int bn = (blockIdx.x >> 5) * BN;   // 8 N-tiles

  const tdm_i32x8 g1A = tdm_desc_g1(BK, BM, HIDDEN, MTOT, HIDDEN);
  const tdm_i32x8 g1B = tdm_desc_g1(BK, BM, HIDDEN, HIDDEN, HIDDEN);
  const _Float16* Xrow = X + (size_t)bm * HIDDEN;
  const _Float16* Wrow = W + (size_t)bn * HIDDEN;

  if (wave == 0) {
    tdm_issue(0u, Xrow, g1A);
    tdm_issue((u32)(2 * TILE_ELTS * 2), Wrow, g1B);
    __builtin_amdgcn_s_wait_tensorcnt(0);
  }
  __syncthreads();

  v8f acc[2][4] = {};
  for (int kt = 0; kt < HIDDEN / BK; ++kt) {
    int cur = kt & 1;
    int nxt = cur ^ 1;
    if (wave == 0 && kt + 1 < HIDDEN / BK) {
      int k0 = (kt + 1) * BK;
      tdm_issue((u32)(nxt * TILE_ELTS * 2), Xrow + k0, g1A);
      tdm_issue((u32)((2 + nxt) * TILE_ELTS * 2), Wrow + k0, g1B);
    }
    // compute current tile from LDS while TDM streams the next one
    const _Float16* Acur = smem + cur * TILE_ELTS;
    const _Float16* Bcur = smem + (2 + cur) * TILE_ELTS;
    v16h a0 = load_a(Acur, mw * 32,      BK, 0, lane);
    v16h a1 = load_a(Acur, mw * 32 + 16, BK, 0, lane);
#pragma unroll
    for (int j = 0; j < 4; ++j) {
      v16h b = load_b(Bcur, nw * 64 + j * 16, BK, 0, lane);
      acc[0][j] = wmma_f16(a0, b, acc[0][j]);
      acc[1][j] = wmma_f16(a1, b, acc[1][j]);
    }
    if (wave == 0 && kt + 1 < HIDDEN / BK) __builtin_amdgcn_s_wait_tensorcnt(0);
    __syncthreads();
  }

  int hi = lane >> 4, cl = lane & 15;
#pragma unroll
  for (int half = 0; half < 2; ++half) {
    int mbase = bm + mw * 32 + half * 16;
#pragma unroll
    for (int j = 0; j < 4; ++j) {
      int col = bn + nw * 64 + j * 16 + cl;
      float bb = bias[col];
#pragma unroll
      for (int r = 0; r < 8; ++r) {
        int m = mbase + hi * 8 + r;
        float v = acc[half][j][r] + bb;
        if (mode == 0) {
          ((float*)Y)[(size_t)m * HIDDEN + col] = v;
        } else {
          int b_ = m >> 11, s = m & (SEQ - 1);
          int h = col >> 6, dh = col & (HDIM - 1);
          if (mode == 3)
            ((_Float16*)Y)[(((size_t)(b_ * HEADS + h)) * HDIM + dh) * SEQ + s] =
                (_Float16)v;
          else
            ((_Float16*)Y)[(((size_t)(b_ * HEADS + h)) * SEQ + s) * HDIM + dh] =
                (_Float16)v;
        }
      }
    }
  }
}

// ---------------- Flash attention with TDM-staged K/V tiles --------------
// Block: one (b,h), 8 waves = 8 query blocks of 16. K-tile 32x64 and V-tile
// (transposed) 64x32 staged to LDS per 32-key step, double-buffered, shared
// by all 8 waves. ctx out: [B, S, H*64] f16.
__global__ void __launch_bounds__(256)
flash_attn(const _Float16* __restrict__ Q, const _Float16* __restrict__ K,
           const _Float16* __restrict__ Vt, _Float16* __restrict__ ctx) {
  // [0,4096): K bufs | [4096,8192): V bufs | [8192,12288): per-wave P staging
  __shared__ __align__(16) _Float16 smem[2 * 32 * HDIM + 2 * HDIM * 32 + 8 * 512];

  int lane = threadIdx.x & 31;
  int wave = threadIdx.x >> 5;
  _Float16* pbuf = smem + 8192 + wave * 512;

  int bh = blockIdx.x >> 4;            // 0..31 = b*16+h
  int qb = (blockIdx.x & 15) * 8 + wave;
  int m0 = qb * 16;
  int hi = lane >> 4, cl = lane & 15;

  const _Float16* Qb = Q  + (size_t)bh * SEQ * HDIM;
  const _Float16* Kb = K  + (size_t)bh * SEQ * HDIM;
  const _Float16* Vb = Vt + (size_t)bh * HDIM * SEQ;

  const tdm_i32x8 g1K = tdm_desc_g1(HDIM, 32, HDIM, SEQ, HDIM);  // 32 keys x 64
  const tdm_i32x8 g1V = tdm_desc_g1(32, HDIM, SEQ, HDIM, SEQ);   // 64 d x 32 keys

  // resident Q fragments (16x64)
  v16h a0 = load_a(Qb, m0, HDIM, 0, lane);
  v16h a1 = load_a(Qb, m0, HDIM, 32, lane);

  float mi[8], li[8];
#pragma unroll
  for (int r = 0; r < 8; ++r) { mi[r] = -1e30f; li[r] = 0.0f; }
  v8f o[4] = {};
  const float scale = 0.125f;  // 1/sqrt(64)

  if (wave == 0) {
    tdm_issue(0u, Kb, g1K);
    tdm_issue(8192u, Vb, g1V);
    __builtin_amdgcn_s_wait_tensorcnt(0);
  }
  __syncthreads();

  for (int kb = 0; kb < SEQ / 32; ++kb) {
    int cur = kb & 1;
    int nxt = cur ^ 1;
    if (wave == 0 && kb + 1 < SEQ / 32) {
      int key0 = (kb + 1) * 32;
      tdm_issue((u32)(nxt * 4096), Kb + (size_t)key0 * HDIM, g1K);
      tdm_issue((u32)(8192 + nxt * 4096), Vb + key0, g1V);
    }

    const _Float16* Kcur = smem + cur * 2048;
    const _Float16* Vcur = smem + 4096 + cur * 2048;

    // ---- scores from LDS K tile: two 16-key tiles, Dh split in two ----
    v8f s0 = {}, s1 = {};
    {
      v16h b0 = load_b(Kcur, 0, HDIM, 0, lane);
      v16h b1 = load_b(Kcur, 0, HDIM, 32, lane);
      s0 = wmma_f16(a0, b0, s0);
      s0 = wmma_f16(a1, b1, s0);
      v16h b2 = load_b(Kcur, 16, HDIM, 0, lane);
      v16h b3 = load_b(Kcur, 16, HDIM, 32, lane);
      s1 = wmma_f16(a0, b2, s1);
      s1 = wmma_f16(a1, b3, s1);
    }

    // ---- online softmax; rows live in (VGPR r, lane-half) ----
#pragma unroll
    for (int r = 0; r < 8; ++r) {
      float x0 = s0[r] * scale;
      float x1 = s1[r] * scale;
      float mx = fmaxf(x0, x1);
#pragma unroll
      for (int off = 1; off < 16; off <<= 1)
        mx = fmaxf(mx, __shfl_xor(mx, off, 32));
      float mnew = fmaxf(mi[r], mx);
      float alpha = __expf(mi[r] - mnew);
      float p0 = __expf(x0 - mnew);
      float p1 = __expf(x1 - mnew);
      float rs = p0 + p1;
#pragma unroll
      for (int off = 1; off < 16; off <<= 1)
        rs += __shfl_xor(rs, off, 32);
      li[r] = li[r] * alpha + rs;
      mi[r] = mnew;
      o[0][r] *= alpha; o[1][r] *= alpha; o[2][r] *= alpha; o[3][r] *= alpha;
      int row = r + hi * 8;
      pbuf[row * 32 + cl]      = (_Float16)p0;
      pbuf[row * 32 + 16 + cl] = (_Float16)p1;
    }

    asm volatile("s_wait_dscnt 0" ::: "memory");  // per-wave LDS round trip

    v16h pf;  // P as A-fragment (16x32)
    {
      const _Float16* p = pbuf + cl * 32 + hi * 8;
      v8h* ph = (v8h*)&pf;
      ph[0] = *(const v8h*)(p);
      ph[1] = *(const v8h*)(p + 16);
    }

    // ---- O += P(16x32) * V(32x64) from LDS V tile ----
#pragma unroll
    for (int j = 0; j < 4; ++j) {
      v16h bv = load_b(Vcur, j * 16, 32, 0, lane);
      o[j] = wmma_f16(pf, bv, o[j]);
    }

    if (wave == 0 && kb + 1 < SEQ / 32) __builtin_amdgcn_s_wait_tensorcnt(0);
    __syncthreads();
  }

  int b_ = bh >> 4, h = bh & 15;
#pragma unroll
  for (int r = 0; r < 8; ++r) {
    float inv = 1.0f / li[r];
    int s = m0 + r + hi * 8;
    size_t base = ((size_t)b_ * SEQ + s) * HIDDEN + h * HDIM;
#pragma unroll
    for (int j = 0; j < 4; ++j)
      ctx[base + j * 16 + cl] = (_Float16)(o[j][r] * inv);
  }
}

extern "C" void kernel_launch(void* const* d_in, const int* in_sizes, int n_in,
                              void* d_out, int out_size, void* d_ws, size_t ws_size,
                              hipStream_t stream) {
  const float* x  = (const float*)d_in[0];
  const float* wq = (const float*)d_in[1];
  const float* bq = (const float*)d_in[2];
  const float* wk = (const float*)d_in[3];
  const float* bk = (const float*)d_in[4];
  const float* wv = (const float*)d_in[5];
  const float* bv = (const float*)d_in[6];
  const float* wo = (const float*)d_in[7];
  const float* bo = (const float*)d_in[8];
  float* out = (float*)d_out;

  const size_t nX = (size_t)MTOT * HIDDEN;     // 4,194,304
  const size_t nW = (size_t)HIDDEN * HIDDEN;   // 1,048,576
  char* ws = (char*)d_ws;
  _Float16* xh   = (_Float16*)(ws);
  _Float16* wqh  = xh  + nX;
  _Float16* wkh  = wqh + nW;
  _Float16* wvh  = wkh + nW;
  _Float16* woh  = wvh + nW;
  _Float16* Qb   = woh + nW;   // [B,H,S,Dh]
  _Float16* Kbf  = Qb  + nX;   // [B,H,S,Dh]
  _Float16* Vtb  = Kbf + nX;   // [B,H,Dh,S]
  _Float16* ctxh = Vtb + nX;   // [B,S,H*Dh]

  cvt_f32_f16<<<(int)((nX + 255) / 256), 256, 0, stream>>>(x, xh, (int)nX);
  cvt_f32_f16<<<(int)((nW + 255) / 256), 256, 0, stream>>>(wq, wqh, (int)nW);
  cvt_f32_f16<<<(int)((nW + 255) / 256), 256, 0, stream>>>(wk, wkh, (int)nW);
  cvt_f32_f16<<<(int)((nW + 255) / 256), 256, 0, stream>>>(wv, wvh, (int)nW);
  cvt_f32_f16<<<(int)((nW + 255) / 256), 256, 0, stream>>>(wo, woh, (int)nW);

  // QKV projections: 256 blocks (32 M-tiles x 8 N-tiles), 128x128 per block
  gemm_f16<<<256, 256, 0, stream>>>(xh, wqh, bq, (void*)Qb,  1);
  gemm_f16<<<256, 256, 0, stream>>>(xh, wkh, bk, (void*)Kbf, 1);
  gemm_f16<<<256, 256, 0, stream>>>(xh, wvh, bv, (void*)Vtb, 3);

  // attention: 512 blocks = 32 (b,h) x 16 query groups (8 waves x 16 queries)
  flash_attn<<<512, 256, 0, stream>>>(Qb, Kbf, Vtb, ctxh);

  // output projection -> fp32
  gemm_f16<<<256, 256, 0, stream>>>(ctxh, woh, bo, (void*)out, 0);
}
